// Model_2997887172798
// MI455X (gfx1250) — matrix operand order repaired and checked by reading, not compiled, not run on hardware
//
#include <hip/hip_runtime.h>
#include <hip/hip_bf16.h>

typedef __attribute__((ext_vector_type(16))) _Float16 v16h;
typedef __attribute__((ext_vector_type(8)))  _Float16 v8h;
typedef __attribute__((ext_vector_type(8)))  float    v8f;

#define DEV __device__ __forceinline__

namespace cfg {
constexpr int BB   = 16;
constexpr int SLEN = 512;
constexpr int ROWS = BB * SLEN;      // 8192 tokens
constexpr int NH   = 4;
constexpr int EMB  = 256;
constexpr int INNER= 512;
constexpr int DHM  = 128;
constexpr int DHS  = 64;
constexpr int PRED = 96;
constexpr int COUT = 21;
constexpr int UPD  = 384;
}
using namespace cfg;

// ---------------------------------------------------------------------------
// WMMA fragment loaders (f32 memory -> f16 fragments), gfx1250 wave32 layouts.
// No bounds checks: every call site uses K % 32 == 0 and 16B-aligned rows,
// so each fragment is a handful of b128 loads + pack, not 32 masked loads.
//
// A 16x32 f16 (ISA 7.12.2): lane l: m = l&15; halves 0..7 -> K=base..base+7,
// halves 8..15 -> K=base+16..base+23, base = (l<16)?0:8.
// ---------------------------------------------------------------------------
DEV v16h load_a_frag_f32(const float* __restrict__ A, int lda, int m0, int k0,
                         int lane) {
  const float* row = A + (size_t)(m0 + (lane & 15)) * lda + k0 +
                     ((lane & 16) ? 8 : 0);
  float4 a0 = *(const float4*)(row);
  float4 a1 = *(const float4*)(row + 4);
  float4 a2 = *(const float4*)(row + 16);
  float4 a3 = *(const float4*)(row + 20);
  v16h a;
  a[0]  = (_Float16)a0.x; a[1]  = (_Float16)a0.y;
  a[2]  = (_Float16)a0.z; a[3]  = (_Float16)a0.w;
  a[4]  = (_Float16)a1.x; a[5]  = (_Float16)a1.y;
  a[6]  = (_Float16)a1.z; a[7]  = (_Float16)a1.w;
  a[8]  = (_Float16)a2.x; a[9]  = (_Float16)a2.y;
  a[10] = (_Float16)a2.z; a[11] = (_Float16)a2.w;
  a[12] = (_Float16)a3.x; a[13] = (_Float16)a3.y;
  a[14] = (_Float16)a3.z; a[15] = (_Float16)a3.w;
  return a;
}

// B 32x16 f16: B[k][n] taken from a row-major W[n][k] (weights stored (N,K)).
// lane l: n = l&15; halves 0..7 -> k = k0+base+j, halves 8..15 -> k0+base+16+(j-8), base = (l<16)?0:8.
DEV v16h load_bT_frag_f32(const float* __restrict__ W, int ldw, int n0, int k0,
                          int lane) {
  const float* row = W + (size_t)(n0 + (lane & 15)) * ldw + k0 +
                     ((lane & 16) ? 8 : 0);
  float4 b0 = *(const float4*)(row);
  float4 b1 = *(const float4*)(row + 4);
  float4 b2 = *(const float4*)(row + 16);
  float4 b3 = *(const float4*)(row + 20);
  v16h b;
  b[0]  = (_Float16)b0.x; b[1]  = (_Float16)b0.y;
  b[2]  = (_Float16)b0.z; b[3]  = (_Float16)b0.w;
  b[4]  = (_Float16)b1.x; b[5]  = (_Float16)b1.y;
  b[6]  = (_Float16)b1.z; b[7]  = (_Float16)b1.w;
  b[8]  = (_Float16)b2.x; b[9]  = (_Float16)b2.y;
  b[10] = (_Float16)b2.z; b[11] = (_Float16)b2.w;
  b[12] = (_Float16)b3.x; b[13] = (_Float16)b3.y;
  b[14] = (_Float16)b3.z; b[15] = (_Float16)b3.w;
  return b;
}

DEV float sigmoidf_(float x) { return 1.f / (1.f + __expf(-x)); }
DEV float siluf_(float x)    { return x * sigmoidf_(x); }
DEV float logsigf_(float x)  { return fminf(x, 0.f) - log1pf(__expf(-fabsf(x))); }

// ---------------------------------------------------------------------------
// WMMA GEMM: C[M,N] (+=) A[M,K] * W[N,K]^T.
// One wave computes a 16x64 tile: A-fragment reused across 4 B-tiles.
// Requires K % 32 == 0, M % 16 == 0, N % 64 == 0 (true at all call sites).
// ---------------------------------------------------------------------------
__global__ __launch_bounds__(32) void gemm_wmma_kernel(
    const float* __restrict__ A, int lda, const float* __restrict__ W, int ldw,
    float* __restrict__ C, int ldc, int Kdim, int addC) {
  int lane = threadIdx.x;
  int m0 = blockIdx.x * 16;
  int n0 = blockIdx.y * 64;
  v8f acc0 = {0.f,0.f,0.f,0.f,0.f,0.f,0.f,0.f};
  v8f acc1 = acc0, acc2 = acc0, acc3 = acc0;
  for (int k0 = 0; k0 < Kdim; k0 += 32) {
    v16h a  = load_a_frag_f32(A, lda, m0, k0, lane);
    v16h b0 = load_bT_frag_f32(W, ldw, n0,      k0, lane);
    v16h b1 = load_bT_frag_f32(W, ldw, n0 + 16, k0, lane);
    v16h b2 = load_bT_frag_f32(W, ldw, n0 + 32, k0, lane);
    v16h b3 = load_bT_frag_f32(W, ldw, n0 + 48, k0, lane);
    acc0 = __builtin_amdgcn_wmma_f32_16x16x32_f16(false, a, false, b0, (short)0, acc0, false, false);
    acc1 = __builtin_amdgcn_wmma_f32_16x16x32_f16(false, a, false, b1, (short)0, acc1, false, false);
    acc2 = __builtin_amdgcn_wmma_f32_16x16x32_f16(false, a, false, b2, (short)0, acc2, false, false);
    acc3 = __builtin_amdgcn_wmma_f32_16x16x32_f16(false, a, false, b3, (short)0, acc3, false, false);
  }
  int rbase = m0 + ((lane & 16) ? 8 : 0);
  int coll  = lane & 15;
  v8f* accs[4] = {&acc0, &acc1, &acc2, &acc3};
#pragma unroll
  for (int t = 0; t < 4; ++t) {
    int col = n0 + t * 16 + coll;
#pragma unroll
    for (int r = 0; r < 8; ++r) {
      size_t off = (size_t)(rbase + r) * ldc + col;
      float v = (*accs[t])[r];
      if (addC) v += C[off];
      C[off] = v;
    }
  }
}

// ---------------------------------------------------------------------------
// Scalar helper kernels
// ---------------------------------------------------------------------------
__global__ void embed_kernel(const float* __restrict__ xe,
                             const float* __restrict__ xm,
                             const float* __restrict__ W,
                             const float* __restrict__ bias,
                             float* __restrict__ x) {
  int idx = blockIdx.x * blockDim.x + threadIdx.x;
  if (idx >= ROWS * EMB) return;
  int e = idx & 255, row = idx >> 8;
  const float* a = xe + (size_t)row * 21;
  const float* m = xm + (size_t)row * 4;
  const float* w = W + (size_t)e * 25;
  float acc = bias[e];
#pragma unroll
  for (int j = 0; j < 21; ++j) acc += a[j] * w[j];
#pragma unroll
  for (int j = 0; j < 4; ++j) acc += m[j] * w[21 + j];
  x[idx] = acc;
}

__global__ void layernorm_kernel(const float* __restrict__ x,
                                 const float* __restrict__ g,
                                 float* __restrict__ y, int rows) {
  int row = blockIdx.x * blockDim.x + threadIdx.x;
  if (row >= rows) return;
  const float* xp = x + (size_t)row * EMB;
  float mu = 0.f, m2 = 0.f;
  for (int c = 0; c < EMB; ++c) { float v = xp[c]; mu += v; m2 += v * v; }
  mu *= (1.f / EMB);
  m2 = m2 * (1.f / EMB) - mu * mu;
  float rs = rsqrtf(m2 + 1e-5f);
  float* yp = y + (size_t)row * EMB;
  for (int c = 0; c < EMB; ++c) yp[c] = (xp[c] - mu) * rs * g[c];
}

__global__ void conv_silu_kernel(const float* __restrict__ x, int ldx,
                                 const float* __restrict__ w,
                                 const float* __restrict__ bias,
                                 float* __restrict__ y, int ldy, int C,
                                 int total) {
  int idx = blockIdx.x * blockDim.x + threadIdx.x;
  if (idx >= total) return;
  int c = idx % C;
  int s = (idx / C) % SLEN;
  int b = idx / (C * SLEN);
  float acc = bias[c];
#pragma unroll
  for (int k = 0; k < 4; ++k) {
    int sp = s - 3 + k;
    if (sp >= 0) acc += x[(size_t)(b * SLEN + sp) * ldx + c] * w[c * 4 + k];
  }
  y[(size_t)(b * SLEN + s) * ldy + c] = siluf_(acc);
}

__global__ void blockdiag_kernel(const float* __restrict__ x, int ldx,
                                 const float* __restrict__ w,
                                 float* __restrict__ y, int ldy, int total) {
  int idx = blockIdx.x * blockDim.x + threadIdx.x;
  if (idx >= total) return;
  int c = idx % INNER;
  int row = idx / INNER;
  int nb = c >> 2, o = c & 3;
  const float* xr = x + (size_t)row * ldx + (nb << 2);
  const float* wr = w + (size_t)(nb * 4 + o) * 4;
  y[(size_t)row * ldy + c] =
      xr[0] * wr[0] + xr[1] * wr[1] + xr[2] * wr[2] + xr[3] * wr[3];
}

__global__ void mlstm_gates_kernel(const float* __restrict__ q,
                                   const float* __restrict__ k,
                                   const float* __restrict__ v,
                                   const float* __restrict__ iw,
                                   const float* __restrict__ ib,
                                   const float* __restrict__ fw,
                                   const float* __restrict__ fb,
                                   float* __restrict__ ipre,
                                   float* __restrict__ fpre) {
  int idx = blockIdx.x * blockDim.x + threadIdx.x;  // b*NH*S
  if (idx >= BB * NH * SLEN) return;
  int s = idx % SLEN;
  int h = (idx / SLEN) % NH;
  int b = idx / (SLEN * NH);
  size_t row = (size_t)(b * SLEN + s) * INNER;
  const float* iwr = iw + (size_t)h * 3 * INNER;
  const float* fwr = fw + (size_t)h * 3 * INNER;
  float si = ib[h], sf = fb[h];
  for (int c = 0; c < INNER; ++c) {
    float xq = q[row + c], xk = k[row + c], xv = v[row + c];
    si += xq * iwr[c] + xk * iwr[INNER + c] + xv * iwr[2 * INNER + c];
    sf += xq * fwr[c] + xk * fwr[INNER + c] + xv * fwr[2 * INNER + c];
  }
  ipre[(size_t)(b * NH + h) * SLEN + s] = si;
  fpre[(size_t)(b * NH + h) * SLEN + s] = sf;
}

__global__ void cumlogsig_kernel(const float* __restrict__ fpre,
                                 float* __restrict__ cf) {
  int idx = blockIdx.x * blockDim.x + threadIdx.x;  // (b,h)
  if (idx >= BB * NH) return;
  const float* f = fpre + (size_t)idx * SLEN;
  float* c = cf + (size_t)idx * (SLEN + 1);
  float acc = 0.f;
  c[0] = 0.f;
  for (int s = 0; s < SLEN; ++s) {
    acc += logsigf_(f[s]);
    c[s + 1] = acc;
  }
}

// ---------------------------------------------------------------------------
// mLSTM stabilized parallel kernel (flash-style, one wave = 16 query rows)
// ---------------------------------------------------------------------------
__global__ __launch_bounds__(32) void mlstm_attn_kernel(
    const float* __restrict__ qg, const float* __restrict__ kg,
    const float* __restrict__ vg, const float* __restrict__ ipre,
    const float* __restrict__ cf, float* __restrict__ out) {
  const int lane = threadIdx.x;
  const int qt = blockIdx.x;
  const int hh = blockIdx.y;
  const int b  = blockIdx.z;
  const int qi = qt * 16;
  const int half = (lane & 16) ? 1 : 0;
  const int ln15 = lane & 15;

  const size_t rowbase = (size_t)b * SLEN * INNER;
  const float* qp = qg + rowbase + hh * DHM;
  const float* kp = kg + rowbase + hh * DHM;
  const float* vp = vg + rowbase + hh * DHM;
  const float* cfp = cf + (size_t)(b * NH + hh) * (SLEN + 1);
  const float* ipp = ipre + (size_t)(b * NH + hh) * SLEN;

  __shared__ _Float16 Pl[16][40];   // padded: rows 80B apart -> 16B-aligned v8h

  v16h qa[4];
#pragma unroll
  for (int c = 0; c < 4; ++c)
    qa[c] = load_a_frag_f32(qp, INNER, qi, c * 32, lane);

  float cfq[8];
#pragma unroll
  for (int r = 0; r < 8; ++r) cfq[r] = cfp[qi + r + half * 8 + 1];

  float mrun[8], srun[8];
#pragma unroll
  for (int r = 0; r < 8; ++r) { mrun[r] = -1e30f; srun[r] = 0.f; }
  v8f acc[8];
#pragma unroll
  for (int g = 0; g < 8; ++g) acc[g] = (v8f){0.f,0.f,0.f,0.f,0.f,0.f,0.f,0.f};

  const float scale = 0.0883883476483184f;  // 1/sqrt(128)

  for (int kt = 0; kt < qi + 16; kt += 32) {
    // scores: Q(16x128) x K^T -> two 16x16 tiles
    v8f s0 = {0.f,0.f,0.f,0.f,0.f,0.f,0.f,0.f};
    v8f s1 = s0;
#pragma unroll
    for (int c = 0; c < 4; ++c) {
      v16h b0 = load_bT_frag_f32(kp, INNER, kt,      c * 32, lane);
      v16h b1 = load_bT_frag_f32(kp, INNER, kt + 16, c * 32, lane);
      s0 = __builtin_amdgcn_wmma_f32_16x16x32_f16(false, qa[c], false, b0,
                                                  (short)0, s0, false, false);
      s1 = __builtin_amdgcn_wmma_f32_16x16x32_f16(false, qa[c], false, b1,
                                                  (short)0, s1, false, false);
    }
    int t0 = kt + ln15, t1 = t0 + 16;
    float cft0 = cfp[t0 + 1], ip0 = ipp[t0];
    float cft1 = cfp[t1 + 1], ip1 = ipp[t1];

#pragma unroll
    for (int r = 0; r < 8; ++r) {
      int i = qi + r + half * 8;
      float l0 = (t0 <= i) ? (cfq[r] - cft0 + ip0) : -1e30f;
      float l1 = (t1 <= i) ? (cfq[r] - cft1 + ip1) : -1e30f;
      float mx = fmaxf(l0, l1);
#pragma unroll
      for (int msk = 8; msk >= 1; msk >>= 1)
        mx = fmaxf(mx, __shfl_xor(mx, msk));
      float mnew = fmaxf(mrun[r], mx);
      float fr = __expf(mrun[r] - mnew);
      mrun[r] = mnew;
      srun[r] *= fr;
#pragma unroll
      for (int g = 0; g < 8; ++g) acc[g][r] *= fr;
      float p0 = s0[r] * scale * __expf(l0 - mnew);
      float p1 = s1[r] * scale * __expf(l1 - mnew);
      srun[r] += p0 + p1;
      Pl[r + half * 8][ln15]      = (_Float16)p0;
      Pl[r + half * 8][16 + ln15] = (_Float16)p1;
    }
    __syncthreads();
    // P as A-fragment (16x32) via vectorized LDS reads
    v16h pa;
    {
      int kb = half ? 8 : 0;
      v8h lo = *(const v8h*)(&Pl[ln15][kb]);
      v8h hi = *(const v8h*)(&Pl[ln15][kb + 16]);
#pragma unroll
      for (int j = 0; j < 8; ++j) { pa[j] = lo[j]; pa[8 + j] = hi[j]; }
    }
    // P x V chunk (32x128)
#pragma unroll
    for (int g = 0; g < 8; ++g) {
      v16h vb;
      int d  = g * 16 + ln15;
      int kb = half ? 8 : 0;
#pragma unroll
      for (int j = 0; j < 16; ++j)
        vb[j] = (_Float16)vp[(size_t)(kt + kb + (j < 8 ? j : j + 8)) * INNER + d];
      acc[g] = __builtin_amdgcn_wmma_f32_16x16x32_f16(false, pa, false, vb,
                                                      (short)0, acc[g], false,
                                                      false);
    }
    __syncthreads();
  }
#pragma unroll
  for (int r = 0; r < 8; ++r) {
    float s = srun[r];
#pragma unroll
    for (int msk = 8; msk >= 1; msk >>= 1) s += __shfl_xor(s, msk);
    float norm = fmaxf(fabsf(s), __expf(-mrun[r])) + 1e-6f;
    float inv = 1.f / norm;
    int i = qi + r + half * 8;
    float* op = out + (size_t)(b * SLEN + i) * INNER + hh * DHM;
#pragma unroll
    for (int g = 0; g < 8; ++g) op[g * 16 + ln15] = acc[g][r] * inv;
  }
}

// per-head norm + skip + silu(z) gate; hbuf updated in place
__global__ void mlstm_postattn_kernel(float* __restrict__ hbuf,
                                      const float* __restrict__ xc,
                                      const float* __restrict__ xi,
                                      const float* __restrict__ onorm,
                                      const float* __restrict__ skip) {
  int idx = blockIdx.x * blockDim.x + threadIdx.x;  // ROWS*NH
  if (idx >= ROWS * NH) return;
  int hd = idx & 3, row = idx >> 2;
  float* hp = hbuf + (size_t)row * INNER + hd * DHM;
  float mu = 0.f, m2 = 0.f;
  for (int d = 0; d < DHM; ++d) { float v = hp[d]; mu += v; m2 += v * v; }
  mu *= (1.f / DHM);
  m2 = m2 * (1.f / DHM) - mu * mu;
  float rs = rsqrtf(m2 + 1e-5f);
  const float* xcr = xc + (size_t)row * INNER;
  const float* zr  = xi + (size_t)row * 1024 + INNER;
  for (int d = 0; d < DHM; ++d) {
    int c = hd * DHM + d;
    float v = (hp[d] - mu) * rs * onorm[c] + skip[c] * xcr[c];
    hp[d] = v * siluf_(zr[c]);
  }
}

// ---------------------------------------------------------------------------
// sLSTM sequential scan: one 64-thread workgroup per (head, batch)
// ---------------------------------------------------------------------------
__global__ __launch_bounds__(64) void slstm_scan_kernel(
    const float* __restrict__ gi, const float* __restrict__ gf,
    const float* __restrict__ gz, const float* __restrict__ go,
    const float* __restrict__ R, const float* __restrict__ bias,
    float* __restrict__ y) {
  int e = threadIdx.x;   // 0..63
  int h = blockIdx.x;    // head
  int b = blockIdx.y;    // batch
  __shared__ float hs[DHS];
  const float* R0 = R + (size_t)(0 * NH + h) * DHS * DHS;
  const float* R1 = R + (size_t)(1 * NH + h) * DHS * DHS;
  const float* R2 = R + (size_t)(2 * NH + h) * DHS * DHS;
  const float* R3 = R + (size_t)(3 * NH + h) * DHS * DHS;
  float bi = bias[(0 * NH + h) * DHS + e];
  float bf = bias[(1 * NH + h) * DHS + e];
  float bz = bias[(2 * NH + h) * DHS + e];
  float bo = bias[(3 * NH + h) * DHS + e];
  float c = 0.f, n = 0.f, m = 0.f;
  hs[e] = 0.f;
  __syncthreads();
  for (int s = 0; s < SLEN; ++s) {
    float ri = 0.f, rf = 0.f, rz = 0.f, ro = 0.f;
    for (int d = 0; d < DHS; ++d) {
      float hp = hs[d];
      ri += hp * R0[d * DHS + e];
      rf += hp * R1[d * DHS + e];
      rz += hp * R2[d * DHS + e];
      ro += hp * R3[d * DHS + e];
    }
    size_t off = (size_t)(b * SLEN + s) * EMB + h * DHS + e;
    float ibar = gi[off] + ri + bi;
    float fbar = gf[off] + rf + bf;
    float zbar = gz[off] + rz + bz;
    float obar = go[off] + ro + bo;
    float lfm = m + logsigf_(fbar);
    float mn = fmaxf(ibar, lfm);
    float ig = __expf(ibar - mn);
    float fg = __expf(lfm - mn);
    c = fg * c + ig * tanhf(zbar);
    n = fg * n + ig;
    m = mn;
    float hv = sigmoidf_(obar) * c / n;
    __syncthreads();
    hs[e] = hv;
    y[off] = hv;
    __syncthreads();
  }
}

__global__ void slstm_ynorm_res_kernel(const float* __restrict__ y,
                                       const float* __restrict__ gn,
                                       float* __restrict__ x) {
  int idx = blockIdx.x * blockDim.x + threadIdx.x;  // ROWS*NH
  if (idx >= ROWS * NH) return;
  int h = idx & 3, row = idx >> 2;
  const float* yp = y + (size_t)row * EMB + h * DHS;
  float mu = 0.f, m2 = 0.f;
  for (int d = 0; d < DHS; ++d) { float v = yp[d]; mu += v; m2 += v * v; }
  mu *= (1.f / DHS);
  m2 = m2 * (1.f / DHS) - mu * mu;
  float rs = rsqrtf(m2 + 1e-5f);
  float* xp = x + (size_t)row * EMB + h * DHS;
  for (int d = 0; d < DHS; ++d)
    xp[d] += (yp[d] - mu) * rs * gn[h * DHS + d];
}

__global__ void gelu_gate_kernel(const float* __restrict__ u,
                                 float* __restrict__ g) {
  int idx = blockIdx.x * blockDim.x + threadIdx.x;  // ROWS*UPD
  if (idx >= ROWS * UPD) return;
  int c = idx % UPD, row = idx / UPD;
  float a  = u[(size_t)row * 768 + c];
  float up = u[(size_t)row * 768 + UPD + c];
  float ge = 0.5f * a * (1.f + erff(a * 0.70710678118654752f));
  g[(size_t)row * UPD + c] = ge * up;
}

__global__ void out_proj_kernel(const float* __restrict__ xn,
                                const float* __restrict__ W,
                                const float* __restrict__ bias,
                                float* __restrict__ out) {
  int idx = blockIdx.x * blockDim.x + threadIdx.x;  // B*PRED*COUT
  if (idx >= BB * PRED * COUT) return;
  int c = idx % COUT;
  int p = (idx / COUT) % PRED;
  int b = idx / (COUT * PRED);
  const float* xp = xn + (size_t)(b * SLEN + (SLEN - PRED) + p) * EMB;
  const float* wr = W + (size_t)c * EMB;
  float acc = bias[c];
  for (int e = 0; e < EMB; ++e) acc += xp[e] * wr[e];
  out[idx] = acc;
}

// ---------------------------------------------------------------------------
// Host orchestration
// ---------------------------------------------------------------------------
static inline void launch_gemm(const float* A, int lda, const float* W, int ldw,
                               float* C, int ldc, int M, int N, int K, int addC,
                               hipStream_t st) {
  dim3 grid(M / 16, N / 64);
  gemm_wmma_kernel<<<grid, 32, 0, st>>>(A, lda, W, ldw, C, ldc, K, addC);
}
static inline dim3 g1(long long total) { return dim3((unsigned)((total + 255) / 256)); }

extern "C" void kernel_launch(void* const* d_in, const int* in_sizes, int n_in,
                              void* d_out, int out_size, void* d_ws,
                              size_t ws_size, hipStream_t stream) {
  (void)in_sizes; (void)out_size; (void)ws_size;
  if (n_in < 103) return;
  auto in = [&](int i) { return (const float*)d_in[i]; };

  int idx = 0;
  const float* x_enc = in(idx++);
  const float* x_mark_enc = in(idx++);
  idx += 2;  // x_dec, x_mark_dec unused by the forward pass

  // params tree, sorted-key order: blocks, emb_W, emb_b, out_W, out_b, post_ln
  // mLSTM block leaves (sorted): Wdown,Wup,conv_b,conv_w,fg_b,fg_w,ig_b,ig_w,
  //                              kw,ln,onorm,qw,skip,vw
  // sLSTM block leaves (sorted): R,b,conv_b,conv_w,ffn_down,ffn_up,gn,gw,ln,ln2
  const float* mp[7][14];
  const float* sp[10] = {};
  for (int blk = 0; blk < 7; ++blk) {
    if (blk == 1) { for (int j = 0; j < 10; ++j) sp[j] = in(idx++); }
    else          { for (int j = 0; j < 14; ++j) mp[blk][j] = in(idx++); }
  }
  const float* emb_W = in(idx++);
  const float* emb_b = in(idx++);
  const float* out_W = in(idx++);
  const float* out_b = in(idx++);
  const float* post_ln = in(idx++);

  // workspace bump allocator (256B aligned blocks)
  char* wp = (char*)d_ws;
  auto alloc = [&](size_t nfloats) {
    float* p = (float*)wp;
    wp += ((nfloats * sizeof(float) + 255) / 256) * 256;
    return p;
  };
  float* x    = alloc((size_t)ROWS * EMB);
  float* xn   = alloc((size_t)ROWS * EMB);
  float* xi   = alloc((size_t)ROWS * 1024);  // Wup out; reused as FFN u / scan y
  float* xc   = alloc((size_t)ROWS * INNER);
  float* qb   = alloc((size_t)ROWS * INNER);
  float* kb   = alloc((size_t)ROWS * INNER);
  float* vb   = alloc((size_t)ROWS * INNER);
  float* hb   = alloc((size_t)ROWS * INNER);
  float* ipre = alloc((size_t)BB * NH * SLEN);
  float* fpre = alloc((size_t)BB * NH * SLEN);
  float* cfb  = alloc((size_t)BB * NH * (SLEN + 1));

  // ---- embedding ----
  embed_kernel<<<g1((long long)ROWS * EMB), 256, 0, stream>>>(
      x_enc, x_mark_enc, emb_W, emb_b, x);

  for (int blk = 0; blk < 7; ++blk) {
    if (blk != 1) {
      const float* const* p = mp[blk];
      layernorm_kernel<<<g1(ROWS), 256, 0, stream>>>(x, p[9], xn, ROWS);
      // Wup: (ROWS,256)x(1024,256)^T -> xi (ld 1024)
      launch_gemm(xn, EMB, p[1], EMB, xi, 1024, ROWS, 1024, EMB, 0, stream);
      // causal conv + silu on xm = xi[:, :512]
      conv_silu_kernel<<<g1((long long)ROWS * INNER), 256, 0, stream>>>(
          xi, 1024, p[3], p[2], xc, INNER, INNER, ROWS * INNER);
      // block-diagonal q,k,v
      blockdiag_kernel<<<g1((long long)ROWS * INNER), 256, 0, stream>>>(
          xc, INNER, p[11], qb, INNER, ROWS * INNER);
      blockdiag_kernel<<<g1((long long)ROWS * INNER), 256, 0, stream>>>(
          xc, INNER, p[8], kb, INNER, ROWS * INNER);
      blockdiag_kernel<<<g1((long long)ROWS * INNER), 256, 0, stream>>>(
          xi, 1024, p[13], vb, INNER, ROWS * INNER);
      // gate preactivations
      mlstm_gates_kernel<<<g1((long long)BB * NH * SLEN), 256, 0, stream>>>(
          qb, kb, vb, p[7], p[6], p[5], p[4], ipre, fpre);
      cumlogsig_kernel<<<1, 64, 0, stream>>>(fpre, cfb);
      // stabilized parallel mLSTM
      dim3 ag(SLEN / 16, NH, BB);
      mlstm_attn_kernel<<<ag, 32, 0, stream>>>(qb, kb, vb, ipre, cfb, hb);
      // head-norm + skip + silu(z) gate
      mlstm_postattn_kernel<<<g1((long long)ROWS * NH), 256, 0, stream>>>(
          hb, xc, xi, p[10], p[12]);
      // Wdown with residual add into x
      launch_gemm(hb, INNER, p[0], INNER, x, EMB, ROWS, EMB, INNER, 1, stream);
    } else {
      // ---- sLSTM block ----
      layernorm_kernel<<<g1(ROWS), 256, 0, stream>>>(x, sp[8], xn, ROWS);
      conv_silu_kernel<<<g1((long long)ROWS * EMB), 256, 0, stream>>>(
          xn, EMB, sp[3], sp[2], xc, EMB, EMB, ROWS * EMB);
      // headwise gate projections: i,f from conv branch; z,o from ln branch
      float* gdst[4] = {qb, kb, vb, hb};
      for (int g = 0; g < 4; ++g) {
        const float* src = (g < 2) ? xc : xn;
        for (int h = 0; h < NH; ++h)
          launch_gemm(src + h * DHS, EMB,
                      sp[7] + (size_t)(g * NH + h) * DHS * DHS, DHS,
                      gdst[g] + h * DHS, EMB, ROWS, DHS, DHS, 0, stream);
      }
      // sequential scan -> y (reuse xi, packed ld=EMB)
      dim3 sg(NH, BB);
      slstm_scan_kernel<<<sg, 64, 0, stream>>>(qb, kb, vb, hb, sp[0], sp[1], xi);
      slstm_ynorm_res_kernel<<<g1((long long)ROWS * NH), 256, 0, stream>>>(
          xi, sp[6], x);
      // gated FFN
      layernorm_kernel<<<g1(ROWS), 256, 0, stream>>>(x, sp[9], xn, ROWS);
      launch_gemm(xn, EMB, sp[5], EMB, xi, 768, ROWS, 768, EMB, 0, stream);
      gelu_gate_kernel<<<g1((long long)ROWS * UPD), 256, 0, stream>>>(xi, qb);
      launch_gemm(qb, UPD, sp[4], UPD, x, EMB, ROWS, EMB, UPD, 1, stream);
    }
  }

  // ---- head ----
  layernorm_kernel<<<g1(ROWS), 256, 0, stream>>>(x, post_ln, xn, ROWS);
  out_proj_kernel<<<g1((long long)BB * PRED * COUT), 256, 0, stream>>>(
      xn, out_W, out_b, (float*)d_out);
}
